// CTLSTM_PP_74938589381017
// MI455X (gfx1250) — compile-verified, matrix operand order checked by reading, not run on hardware
//
#include <hip/hip_runtime.h>
#include <hip/hip_bf16.h>
#include <math.h>

// ---------------- problem constants ----------------
#define BSZ   128
#define LSEQ  512
#define HDIM  256
#define EDIM  64
#define NGATE 7
#define KV    (EDIM + HDIM)   // 320 = layer-1 K
#define STEPS (LSEQ - 1)      // 511 sequential steps
#define BT    32              // batch rows per workgroup (2 WMMA M-tiles)
#define NMT   (BT / 16)       // 2 M-tiles
#define NBLK  (BSZ / BT)      // 4 workgroups, fully independent
#define KT1   (KV / 32)       // 10 k-steps, layer 1
#define KT2   (HDIM / 32)     // 8  k-steps, layer 2
#define NT_PER_WAVE 2         // each wave owns 32 columns of H (2 n-tiles)

typedef __attribute__((ext_vector_type(16))) __bf16 v16bf;
typedef __attribute__((ext_vector_type(8)))  __bf16 v8bf;
typedef __attribute__((ext_vector_type(8)))  float  v8f;

__device__ __forceinline__ float sigf(float x)      { return 1.f / (1.f + expf(-x)); }
__device__ __forceinline__ float softplusf(float x) { return x > 20.f ? x : log1pf(expf(x)); }

// Build a 16x32 bf16 A-fragment from a row-major [16][rowStride] bf16 LDS matrix.
// ISA 16-bit A layout: lane m (m=lane&15) holds row M=m; low lanes hold
// K = {k0+0..7, k0+16..23}, high lanes hold K = {k0+8..15, k0+24..31}.
__device__ __forceinline__ v16bf afrag(const __bf16* base, int rowStride, int k0, int lane) {
  int m   = lane & 15;
  int off = (lane & 16) ? 8 : 0;
  const __bf16* p = base + m * rowStride + k0 + off;
  v8bf lo = *(const v8bf*)(p);
  v8bf hi = *(const v8bf*)(p + 16);
  return __builtin_shufflevector(lo, hi, 0,1,2,3,4,5,6,7,8,9,10,11,12,13,14,15);
}

// ---------------- prep kernel 1: collapse embedding MLPs to lookup tables ----
// one_hot @ W is a row gather, so te/se only take 4 / 3 distinct values.
__global__ void prep_tables(const float* __restrict__ Wte1, const float* __restrict__ bte1,
                            const float* __restrict__ Wte2, const float* __restrict__ bte2,
                            const float* __restrict__ Wse1, const float* __restrict__ bse1,
                            const float* __restrict__ Wse2, const float* __restrict__ bse2,
                            float* __restrict__ te_tab, float* __restrict__ se_tab) {
  __shared__ float tm[4][EDIM];
  __shared__ float sm[3][EDIM];
  int j = threadIdx.x;  // 64 threads
  for (int e = 0; e < 4; ++e) tm[e][j] = tanhf(Wte1[e * EDIM + j] + bte1[j]);
  for (int s = 0; s < 3; ++s) sm[s][j] = tanhf(Wse1[s * EDIM + j] + bse1[j]);
  __syncthreads();
  for (int e = 0; e < 4; ++e) {
    float a = bte2[j];
    for (int k = 0; k < EDIM; ++k) a += tm[e][k] * Wte2[k * EDIM + j];
    te_tab[e * EDIM + j] = tanhf(a);
  }
  for (int s = 0; s < 3; ++s) {
    float a = bse2[j];
    for (int k = 0; k < EDIM; ++k) a += sm[s][k] * Wse2[k * EDIM + j];
    se_tab[s * EDIM + j] = tanhf(a);
  }
}

// ---------------- prep kernel 2: swizzle f32 weights -> bf16 WMMA B-fragments -
// B-fragment layout (mirror of ISA 16-bit A layout, transposed): for a 32x16
// (KxN) block, lane = nn + 16*((kk>>3)&1), slot = (kk&7) + 8*(kk>>4).
// Block stored as [lane][slot] so the hot loop does one contiguous 32B load/lane.
__global__ void prep_w(const float* __restrict__ W, __bf16* __restrict__ Ws,
                       int Kdim, int total) {
  int idx = blockIdx.x * blockDim.x + threadIdx.x;
  if (idx >= total) return;
  int n  = idx % HDIM;
  int k  = (idx / HDIM) % Kdim;
  int g  = idx / (HDIM * Kdim);
  int kt = k >> 5, kk = k & 31, nt = n >> 4, nn = n & 15;
  int lane = nn + (((kk >> 3) & 1) << 4);
  int slot = (kk & 7) + ((kk >> 4) << 3);
  int KT   = Kdim >> 5;
  size_t pos = (((size_t)(g * (HDIM / 16) + nt) * KT + kt) * 32 + lane) * 16 + slot;
  Ws[pos] = (__bf16)W[idx];
}

// ---------------- main persistent recurrence kernel ------------------------
// One workgroup (8 waves / 256 threads) per 32 batch rows. Wave w owns H
// columns [32w, 32w+32) for ALL batch rows. The recurrence is L2-weight-
// stream bound (2 MB of bf16 weights re-read per step), so each B-fragment
// loaded from L2 is applied to both 16-row M-tiles (2 WMMAs per 32B/lane
// stream element). Layer-2 accumulators for all 7 gates x 2 n-tiles x
// 2 M-tiles live in VGPRs so the gate-mix stage never leaves registers.
__global__ __launch_bounds__(256)
void ctlstm_main(const int*   __restrict__ event,
                 const float* __restrict__ time_step,
                 const int*   __restrict__ state,
                 const float* __restrict__ te_tab,
                 const float* __restrict__ se_tab,
                 const __bf16* __restrict__ W1s,
                 const __bf16* __restrict__ W2s,
                 const float* __restrict__ bg1,
                 const float* __restrict__ bg2,
                 const float* __restrict__ Wint,
                 const float* __restrict__ bint,
                 float* __restrict__ out) {
  __shared__ __align__(32) __bf16 sVA[BT][KV];           // [x_t | h] bf16, layer-1 A
  __shared__ __align__(32) __bf16 sAct[NGATE][BT][HDIM]; // tanh(g1) bf16, layer-2 A
  __shared__ float sDt[BT];

  const int tid    = threadIdx.x;
  const int wave   = tid >> 5;
  const int lane   = tid & 31;
  const int laneN  = lane & 15;
  const int mBase  = (lane & 16) ? 8 : 0;   // C/D layout: VGPR j -> row M = j + mBase
  const int b0     = blockIdx.x * BT;

  float* o_intens = out;
  float* o_cells  = o_intens + (size_t)BSZ * STEPS * 4;
  float* o_ctg    = o_cells  + (size_t)BSZ * STEPS * HDIM;
  float* o_outg   = o_ctg    + (size_t)BSZ * STEPS * HDIM;
  float* o_decay  = o_outg   + (size_t)BSZ * STEPS * HDIM;

  // carried cell states: (mt, l, j) <-> row mt*16+j+mBase, col 32*wave+16*l+laneN
  float cC[NMT][NT_PER_WAVE][8];
  float cT[NMT][NT_PER_WAVE][8];
  #pragma unroll
  for (int mt = 0; mt < NMT; ++mt)
    #pragma unroll
    for (int l = 0; l < NT_PER_WAVE; ++l)
      #pragma unroll
      for (int j = 0; j < 8; ++j) { cC[mt][l][j] = 0.f; cT[mt][l][j] = 0.f; }

  // h(0) = 0
  for (int i = tid; i < BT * HDIM; i += 256)
    sVA[i >> 8][EDIM + (i & 255)] = (__bf16)0.f;

  for (int t = 0; t < STEPS; ++t) {
    // ---- phase 0: stage x_t (table lookup) and dt -----------------------
    {
      int r = tid >> 3, j8 = (tid & 7) * 8;
      int b = b0 + r;
      const float* tp = te_tab + event[b * LSEQ + t] * EDIM;
      const float* sp = se_tab + state[b * LSEQ + t] * EDIM;
      #pragma unroll
      for (int j = 0; j < 8; ++j)
        sVA[r][j8 + j] = (__bf16)(tp[j8 + j] + sp[j8 + j]);
      if (tid < BT) sDt[tid] = time_step[(b0 + tid) * LSEQ + t + 1];
    }
    __syncthreads();

    // ---- phase 1: g1 = tanh([x|h] @ Wg1 + bg1) --------------------------
    // each B-fragment feeds both M-tiles
    for (int g = 0; g < NGATE; ++g) {
      #pragma unroll
      for (int l = 0; l < NT_PER_WAVE; ++l) {
        int ntg = wave * NT_PER_WAVE + l;
        int col = ntg * 16 + laneN;
        v8f acc[NMT];
        #pragma unroll
        for (int mt = 0; mt < NMT; ++mt) acc[mt] = (v8f){};
        const __bf16* wb = W1s + ((size_t)(g * (HDIM / 16) + ntg) * KT1) * 512 + lane * 16;
        #pragma unroll
        for (int kt = 0; kt < KT1; ++kt) {
          v16bf b = *(const v16bf*)(wb + kt * 512);
          #pragma unroll
          for (int mt = 0; mt < NMT; ++mt) {
            v16bf a = afrag(&sVA[mt * 16][0], KV, kt * 32, lane);
            acc[mt] = __builtin_amdgcn_wmma_f32_16x16x32_bf16(false, a, false, b,
                                                              (short)0, acc[mt], false, false);
          }
        }
        float bias = bg1[g * HDIM + col];
        #pragma unroll
        for (int mt = 0; mt < NMT; ++mt)
          #pragma unroll
          for (int j = 0; j < 8; ++j)
            sAct[g][mt * 16 + j + mBase][col] = (__bf16)tanhf(acc[mt][j] + bias);
      }
    }
    __syncthreads();

    // ---- phase 2: g2 = tanh(g1) @ Wg2, register-resident for all 7 gates
    v8f acc2[NMT][NGATE][NT_PER_WAVE];
    #pragma unroll
    for (int g = 0; g < NGATE; ++g) {
      #pragma unroll
      for (int l = 0; l < NT_PER_WAVE; ++l) {
        int ntg = wave * NT_PER_WAVE + l;
        #pragma unroll
        for (int mt = 0; mt < NMT; ++mt) acc2[mt][g][l] = (v8f){};
        const __bf16* wb = W2s + ((size_t)(g * (HDIM / 16) + ntg) * KT2) * 512 + lane * 16;
        #pragma unroll
        for (int kt = 0; kt < KT2; ++kt) {
          v16bf b = *(const v16bf*)(wb + kt * 512);
          #pragma unroll
          for (int mt = 0; mt < NMT; ++mt) {
            v16bf a = afrag(&sAct[g][mt * 16][0], HDIM, kt * 32, lane);
            acc2[mt][g][l] = __builtin_amdgcn_wmma_f32_16x16x32_bf16(false, a, false, b,
                                                                     (short)0, acc2[mt][g][l],
                                                                     false, false);
          }
        }
      }
    }

    // ---- phase 3: gate mix, cell decay, outputs (pure VGPR math) --------
    #pragma unroll
    for (int mt = 0; mt < NMT; ++mt) {
      #pragma unroll
      for (int l = 0; l < NT_PER_WAVE; ++l) {
        int col = (wave * NT_PER_WAVE + l) * 16 + laneN;
        float b2[NGATE];
        #pragma unroll
        for (int g = 0; g < NGATE; ++g) b2[g] = bg2[g * HDIM + col];
        #pragma unroll
        for (int j = 0; j < 8; ++j) {
          int m = mt * 16 + j + mBase;
          float inpt  = sigf(acc2[mt][0][l][j] + b2[0]);
          float forg  = sigf(acc2[mt][1][l][j] + b2[1]);
          float outg  = sigf(acc2[mt][2][l][j] + b2[2]);
          float intg  = sigf(acc2[mt][3][l][j] + b2[3]);
          float fgtg  = sigf(acc2[mt][4][l][j] + b2[4]);
          float z     = tanhf(acc2[mt][5][l][j] + b2[5]);
          float dec   = softplusf(acc2[mt][6][l][j] + b2[6]);
          float ci    = forg * cC[mt][l][j] + inpt * z;
          float ctn   = fgtg * cT[mt][l][j] + intg * z;
          float cn    = ctn + (ci - ctn) * expf(-dec * sDt[m]);
          float hn    = outg * tanhf(cn);
          cC[mt][l][j] = cn;
          cT[mt][l][j] = ctn;
          sVA[m][EDIM + col] = (__bf16)hn;   // feeds next step's layer-1 A + intensity head
          size_t oi = ((size_t)(b0 + m) * STEPS + t) * HDIM + col;
          o_cells[oi] = ci;
          o_ctg[oi]   = ctn;
          o_outg[oi]  = outg;
          o_decay[oi] = dec;
        }
      }
    }
    __syncthreads();

    // ---- phase 4: intensity head, softplus(h @ Wint + bint) -------------
    if (tid < BT * 4) {
      int r = tid >> 2, e = tid & 3;
      float s = bint[e];
      for (int k = 0; k < HDIM; ++k) s += (float)sVA[r][EDIM + k] * Wint[k * 4 + e];
      o_intens[((size_t)(b0 + r) * STEPS + t) * 4 + e] = softplusf(s);
    }
    // no barrier needed: next phase-0 touches only sVA[:, :EDIM] / sDt, and
    // the post-phase-0 barrier orders everything before phase-1/3 touch h.
  }
}

// ---------------- launch ----------------------------------------------------
extern "C" void kernel_launch(void* const* d_in, const int* in_sizes, int n_in,
                              void* d_out, int out_size, void* d_ws, size_t ws_size,
                              hipStream_t stream) {
  const int*   event     = (const int*)  d_in[0];
  const float* time_step = (const float*)d_in[1];
  const int*   state     = (const int*)  d_in[2];
  const float* Wte1 = (const float*)d_in[3];
  const float* bte1 = (const float*)d_in[4];
  const float* Wte2 = (const float*)d_in[5];
  const float* bte2 = (const float*)d_in[6];
  const float* Wse1 = (const float*)d_in[7];
  const float* bse1 = (const float*)d_in[8];
  const float* Wse2 = (const float*)d_in[9];
  const float* bse2 = (const float*)d_in[10];
  const float* Wg1  = (const float*)d_in[11];
  const float* bg1  = (const float*)d_in[12];
  const float* Wg2  = (const float*)d_in[13];
  const float* bg2  = (const float*)d_in[14];
  const float* Wint = (const float*)d_in[15];
  const float* bint = (const float*)d_in[16];

  // workspace layout
  float*  te_tab = (float*)d_ws;                         // 4*64 f32
  float*  se_tab = te_tab + 256;                         // 3*64 f32
  __bf16* W1s    = (__bf16*)((char*)d_ws + 4096);        // 7*16*10*512 bf16
  __bf16* W2s    = W1s + (size_t)NGATE * 16 * KT1 * 512; // 7*16*8*512  bf16

  prep_tables<<<1, 64, 0, stream>>>(Wte1, bte1, Wte2, bte2,
                                    Wse1, bse1, Wse2, bse2, te_tab, se_tab);
  {
    int total1 = NGATE * KV * HDIM;    // 573440
    prep_w<<<(total1 + 255) / 256, 256, 0, stream>>>(Wg1, W1s, KV, total1);
    int total2 = NGATE * HDIM * HDIM;  // 458752
    prep_w<<<(total2 + 255) / 256, 256, 0, stream>>>(Wg2, W2s, HDIM, total2);
  }
  ctlstm_main<<<NBLK, 256, 0, stream>>>(event, time_step, state,
                                        te_tab, se_tab, W1s, W2s,
                                        bg1, bg2, Wint, bint, (float*)d_out);
}
